// Attention_5299989643309
// MI455X (gfx1250) — matrix operand order repaired and checked by reading, not compiled, not run on hardware
//
#include <hip/hip_runtime.h>

#define B_DIM   16
#define S_DIM   2048
#define D_DIM   1024
#define OUT_DIM 1024
#define K3      (3 * D_DIM)
#define M_TILE  32
#define LDS_PITCH 40   // 32 bf16 + 8 pad halfs per row

typedef __attribute__((ext_vector_type(16))) __bf16 v16bf;
typedef __attribute__((ext_vector_type(8)))  float  v8f;

union AFrag { v16bf v; uint4 q[2]; };
union BFrag { v16bf v; uint4 q[2]; __bf16 h[16]; };

// Swizzled-W geometry: tile = 16(o) x 32(k) bf16 = 512 halfs = 1 KB.
// tile id t = ((mat*32 + kt)*64 + ot); inside a tile, lane L's 32B chunk is
// exactly its WMMA B-fragment: o = ot*16 + (L&15), k = kt*32 + (L>>4)*16 + 0..15.
#define WSW_HALFS (2L * 32 * 64 * 512)   // 4 MB

// Issue one B fragment (2 x b128) via volatile asm: invisible to the waitcnt
// pass and un-sinkable, so the loads genuinely stay in flight under the WMMAs.
#define LOAD_BFRAG(frag, ptr)                                          \
    asm volatile("global_load_b128 %0, %2, off\n\t"                    \
                 "global_load_b128 %1, %2, off offset:16"              \
                 : "=&v"((frag).q[0]), "=&v"((frag).q[1])              \
                 : "v"(ptr) : "memory")

// ---------------------------------------------------------------------------
// Kernel 0: W (f32) -> fragment-swizzled bf16 copy (hidden & encoder slices)
// ---------------------------------------------------------------------------
__global__ __launch_bounds__(256)
void wswizzle_kernel(const float* __restrict__ W, __bf16* __restrict__ Wsw)
{
    const int tile = blockIdx.x * 8 + (threadIdx.x >> 5);
    const int lane = threadIdx.x & 31;
    const int ot   = tile & 63;
    const int kt   = (tile >> 6) & 31;
    const int mat  = tile >> 11;
    const int o    = ot * 16 + (lane & 15);
    const int kbase = (mat ? 2 * D_DIM : 0) + kt * 32 + (lane >> 4) * 16;

    const float4* wp = (const float4*)(W + (long)o * K3 + kbase);
    BFrag fb;
#pragma unroll
    for (int q = 0; q < 4; ++q) {
        float4 t = wp[q];
        fb.h[4*q+0] = (__bf16)t.x; fb.h[4*q+1] = (__bf16)t.y;
        fb.h[4*q+2] = (__bf16)t.z; fb.h[4*q+3] = (__bf16)t.w;
    }
    uint4* dst = (uint4*)(Wsw + (long)tile * 512) + lane * 2;
    dst[0] = fb.q[0];
    dst[1] = fb.q[1];
}

// ---------------------------------------------------------------------------
// Kernel 1: cbias[b,o] = bias[o] + sum_d c_t[b,d] * W[o, D + d]
// ---------------------------------------------------------------------------
__global__ __launch_bounds__(256)
void cbias_kernel(const float* __restrict__ c_t, const float* __restrict__ W,
                  const float* __restrict__ bias, float* __restrict__ cbias)
{
    int idx = blockIdx.x * 256 + threadIdx.x;
    int b = idx >> 10;
    int o = idx & (OUT_DIM - 1);
    const float4* wr = (const float4*)(W + (long)o * K3 + D_DIM);
    const float4* cr = (const float4*)(c_t + (long)b * D_DIM);
    float s = bias[o];
    for (int d = 0; d < D_DIM / 4; ++d) {
        float4 w4 = wr[d], c4 = cr[d];
        s += w4.x * c4.x + w4.y * c4.y + w4.z * c4.z + w4.w * c4.w;
    }
    cbias[idx] = s;
}

// ---------------------------------------------------------------------------
// Kernel 2: fused dual-GEMM + cbias + relu + dot(v) -> logits[b,s]
// ---------------------------------------------------------------------------
__global__ __launch_bounds__(256)
void attn_energy_kernel(const float* __restrict__ hidden,
                        const float* __restrict__ enc,
                        const __bf16* __restrict__ Wsw,
                        const float* __restrict__ cbias,
                        const float* __restrict__ v,
                        float* __restrict__ logits)
{
    __shared__ alignas(16) __bf16 ldsA[2][2][M_TILE][LDS_PITCH]; // [buf][mat][row][k]
    __shared__ float rowsum[M_TILE];

    const int wg   = blockIdx.x;
    const int b    = wg >> 6;                 // S/M_TILE = 64
    const int s0   = (wg & 63) * M_TILE;
    const int tid  = threadIdx.x;
    const int wave = tid >> 5;
    const int lane = tid & 31;
    const int lmod = lane & 15;
    const int lhi  = lane >> 4;

    if (tid < M_TILE) rowsum[tid] = 0.f;

    float vv[8], cb[8];
#pragma unroll
    for (int nt = 0; nt < 8; ++nt) {
        int n = wave * 128 + nt * 16 + lmod;
        vv[nt] = v[n];
        cb[nt] = cbias[(long)b * OUT_DIM + n];
    }

    v8f zero = {};
    v8f acc[2][8];
#pragma unroll
    for (int mt = 0; mt < 2; ++mt)
#pragma unroll
        for (int nt = 0; nt < 8; ++nt)
            acc[mt][nt] = zero;

    const long rowBase = ((long)b * S_DIM + s0) * D_DIM;
    const int srow = tid >> 3;
    const int skq  = (tid & 7) * 4;

    auto stage = [&](int buf, int kk) {
        const float4 hv = *(const float4*)(hidden + rowBase + (long)srow * D_DIM + kk + skq);
        const float4 ev = *(const float4*)(enc    + rowBase + (long)srow * D_DIM + kk + skq);
        union { __bf16 h[4]; uint2 u; } p;
        p.h[0] = (__bf16)hv.x; p.h[1] = (__bf16)hv.y;
        p.h[2] = (__bf16)hv.z; p.h[3] = (__bf16)hv.w;
        *(uint2*)&ldsA[buf][0][srow][skq] = p.u;
        p.h[0] = (__bf16)ev.x; p.h[1] = (__bf16)ev.y;
        p.h[2] = (__bf16)ev.z; p.h[3] = (__bf16)ev.w;
        *(uint2*)&ldsA[buf][1][srow][skq] = p.u;
    };

    stage(0, 0);
    __syncthreads();

    int cur = 0;
    for (int kk = 0; kk < D_DIM; kk += 32) {
        // Stage next A tile first (plain C): the compiler's own loadcnt
        // bookkeeping for these loads completes before the asm pipeline below.
        if (kk + 32 < D_DIM) stage(cur ^ 1, kk + 32);

        const int kt = kk >> 5;
        const uint4* bp0 = (const uint4*)(Wsw + (long)((kt * 64) + wave * 8) * 512) + lane * 2;
        const uint4* bp1 = (const uint4*)(Wsw + (long)(((32 + kt) * 64) + wave * 8) * 512) + lane * 2;

        // 4-deep B pipeline, prefetch distance 3, manual waitcnt control.
        BFrag fb[4];
#pragma unroll
        for (int p = 0; p < 3; ++p)               // prologue: steps 0..2 (mat 0)
            LOAD_BFRAG(fb[p], bp0 + p * 64);

#pragma unroll
        for (int mat = 0; mat < 2; ++mat) {
            // A fragments for this matrix (halfs[0..15]=K lhi*16+0..15)
            AFrag a0, a1;
            {
                const int c0 = lhi * 16;
                const __bf16* p0 = &ldsA[cur][mat][lmod][c0];
                a0.q[0] = *(const uint4*)p0;
                a0.q[1] = *(const uint4*)(p0 + 8);
                const __bf16* p1 = &ldsA[cur][mat][16 + lmod][c0];
                a1.q[0] = *(const uint4*)p1;
                a1.q[1] = *(const uint4*)(p1 + 8);
            }
#pragma unroll
            for (int nt = 0; nt < 8; ++nt) {
                const int step = mat * 8 + nt;
                if (step + 3 < 16) {
                    const int s2 = step + 3;
                    LOAD_BFRAG(fb[s2 & 3], ((s2 >> 3) ? bp1 : bp0) + (s2 & 7) * 64);
                }
                // Wait only until the current fragment has landed; LOADcnt
                // retires in order, so <= 2*(frags still in flight) suffices.
                if (step < 13)
                    asm volatile("s_wait_loadcnt 0x6" : "+v"(fb[step & 3].v));
                else if (step == 13)
                    asm volatile("s_wait_loadcnt 0x4" : "+v"(fb[step & 3].v));
                else if (step == 14)
                    asm volatile("s_wait_loadcnt 0x2" : "+v"(fb[step & 3].v));
                else
                    asm volatile("s_wait_loadcnt 0x0" : "+v"(fb[step & 3].v));

                acc[0][nt] = __builtin_amdgcn_wmma_f32_16x16x32_bf16(
                    false, a0.v, false, fb[step & 3].v, (short)0, acc[0][nt], false, false);
                acc[1][nt] = __builtin_amdgcn_wmma_f32_16x16x32_bf16(
                    false, a1.v, false, fb[step & 3].v, (short)0, acc[1][nt], false, false);
            }
        }
        __syncthreads();
        cur ^= 1;
    }

    // epilogue: relu(acc + cbias) * v[n], reduce over n
    float part[2][8];
#pragma unroll
    for (int mt = 0; mt < 2; ++mt)
#pragma unroll
        for (int r = 0; r < 8; ++r)
            part[mt][r] = 0.f;

#pragma unroll
    for (int nt = 0; nt < 8; ++nt) {
#pragma unroll
        for (int mt = 0; mt < 2; ++mt) {
#pragma unroll
            for (int r = 0; r < 8; ++r) {
                float e = acc[mt][nt][r] + cb[nt];
                e = e > 0.f ? e : 0.f;
                part[mt][r] += e * vv[nt];
            }
        }
    }
#pragma unroll
    for (int mt = 0; mt < 2; ++mt)
#pragma unroll
        for (int r = 0; r < 8; ++r)
            atomicAdd(&rowsum[mt * 16 + lhi * 8 + r], part[mt][r]);
    __syncthreads();

    if (tid < M_TILE)
        logits[(long)b * S_DIM + s0 + tid] = rowsum[tid];
}

// ---------------------------------------------------------------------------
// Kernel 3: softmax over S per batch row
// ---------------------------------------------------------------------------
__global__ __launch_bounds__(256)
void softmax_kernel(const float* __restrict__ logits, float* __restrict__ out)
{
    const int b = blockIdx.x;
    const int tid = threadIdx.x;
    __shared__ float red[256];

    float l[8];
    float m = -3.4e38f;
#pragma unroll
    for (int i = 0; i < 8; ++i) {
        l[i] = logits[(long)b * S_DIM + tid + i * 256];
        m = fmaxf(m, l[i]);
    }
    red[tid] = m; __syncthreads();
    for (int s = 128; s > 0; s >>= 1) {
        if (tid < s) red[tid] = fmaxf(red[tid], red[tid + s]);
        __syncthreads();
    }
    m = red[0]; __syncthreads();

    float sum = 0.f;
#pragma unroll
    for (int i = 0; i < 8; ++i) { l[i] = __expf(l[i] - m); sum += l[i]; }
    red[tid] = sum; __syncthreads();
    for (int s = 128; s > 0; s >>= 1) {
        if (tid < s) red[tid] += red[tid + s];
        __syncthreads();
    }
    float inv = 1.f / red[0];
#pragma unroll
    for (int i = 0; i < 8; ++i)
        out[(long)b * S_DIM + tid + i * 256] = l[i] * inv;
}

// ---------------------------------------------------------------------------
extern "C" void kernel_launch(void* const* d_in, const int* in_sizes, int n_in,
                              void* d_out, int out_size, void* d_ws, size_t ws_size,
                              hipStream_t stream)
{
    const float* hidden = (const float*)d_in[0];
    const float* enc    = (const float*)d_in[1];
    const float* c_t    = (const float*)d_in[2];
    const float* W      = (const float*)d_in[3];
    const float* bias   = (const float*)d_in[4];
    const float* v      = (const float*)d_in[5];
    float* out = (float*)d_out;

    __bf16* Wsw   = (__bf16*)d_ws;                       // 4 MB swizzled bf16 W
    float*  cbias = (float*)((char*)d_ws + WSW_HALFS * sizeof(__bf16));
    float*  logits = cbias + B_DIM * OUT_DIM;

    wswizzle_kernel<<<(2 * 32 * 64) / 8, 256, 0, stream>>>(W, Wsw);
    cbias_kernel<<<(B_DIM * OUT_DIM) / 256, 256, 0, stream>>>(c_t, W, bias, cbias);
    attn_energy_kernel<<<B_DIM * (S_DIM / M_TILE), 256, 0, stream>>>(
        hidden, enc, Wsw, cbias, v, logits);
    softmax_kernel<<<B_DIM, 256, 0, stream>>>(logits, out);
}
